// GIN_50483045597680
// MI455X (gfx1250) — compile-verified, matrix operand order checked by reading, not run on hardware
//
#include <hip/hip_runtime.h>
#include <hip/hip_bf16.h>

typedef __attribute__((ext_vector_type(16))) _Float16 v16h;
typedef __attribute__((ext_vector_type(8)))  _Float16 f16x8;
typedef __attribute__((ext_vector_type(4)))  _Float16 f16x4;
typedef __attribute__((ext_vector_type(8)))  float    v8f;

#define HDIM 128
#define TM   128           // rows per block tile
#define LDA  136           // padded LDS row stride (f16 units): 272B, breaks bank conflicts, keeps 16B align

// Non-returning HW float atomic: GLOBAL_ATOMIC_ADD_F32 at device scope.
// Guarantees the single-instruction L2-side RMW (tracked on STOREcnt) instead
// of a possible CAS-loop expansion of the generic atomic builtin.
__device__ __forceinline__ void atomicAddF32(float* p, float v) {
    asm volatile("global_atomic_add_f32 %0, %1, off scope:SCOPE_DEV"
                 :
                 : "v"(p), "v"(v)
                 : "memory");
}

__device__ __forceinline__ v16h cat8(f16x8 lo, f16x8 hi) {
    return __builtin_shufflevector(lo, hi, 0,1,2,3,4,5,6,7,8,9,10,11,12,13,14,15);
}

// ---------------------------------------------------------------------------
// Convert one fp32 weight matrix [K=128][N=128] to f16 transposed [N][K].
__global__ __launch_bounds__(256) void prep_w(const float* __restrict__ w,
                                              _Float16* __restrict__ bt) {
    int i = blockIdx.x * 256 + threadIdx.x;      // 64 blocks -> 16384 elems
    int k = i >> 7, n = i & 127;
    bt[n * HDIM + k] = (_Float16)w[k * HDIM + n];
}

// agg = src (vectorized copy, float4)
__global__ __launch_bounds__(256) void copy_kernel(float4* __restrict__ dst,
                                                   const float4* __restrict__ src, int n4) {
    int i = blockIdx.x * 256 + threadIdx.x;
    if (i < n4) dst[i] = src[i];
}

// agg[dst[e]] += h[src[e]] : one wave per edge (coalesced 512B gather),
// 4 floats per lane -> 4 fire-and-forget global_atomic_add_f32.
__global__ __launch_bounds__(256) void edge_scatter(float* __restrict__ agg,
                                                    const float* __restrict__ h,
                                                    const int* __restrict__ esrc,
                                                    const int* __restrict__ edst, int E) {
    long long t = (long long)blockIdx.x * 256 + threadIdx.x;
    int e = (int)(t >> 5);
    if (e >= E) return;
    int f0 = ((int)t & 31) * 4;
    int s = esrc[e], d = edst[e];
    float4 v = *(const float4*)&h[s * HDIM + f0];
    float* p = &agg[d * HDIM + f0];
    atomicAddF32(p + 0, v.x);
    atomicAddF32(p + 1, v.y);
    atomicAddF32(p + 2, v.z);
    atomicAddF32(p + 3, v.w);
}

__global__ __launch_bounds__(256) void zero_stats(float* __restrict__ stats) {
    stats[threadIdx.x] = 0.0f;   // 256 floats: colsum[128] ++ colsumsq[128]
}

// scale/shift for fused BN apply: y = z*scale + shift
__global__ __launch_bounds__(128) void bn_finalize(const float* __restrict__ colsum,
                                                   const float* __restrict__ colsq,
                                                   const float* __restrict__ gamma,
                                                   const float* __restrict__ beta,
                                                   float* __restrict__ scale,
                                                   float* __restrict__ shift, float invN) {
    int f = threadIdx.x;
    float mu  = colsum[f] * invN;
    float var = colsq[f] * invN - mu * mu;
    float s   = gamma[f] * rsqrtf(var + 1e-5f);
    scale[f] = s;
    shift[f] = beta[f] - mu * s;
}

// ---------------------------------------------------------------------------
// WMMA GEMM: out[M,128] = A[M,128] @ B[128,128] (+bias, epilogue per MODE)
// MODE 0: A raw fp32; out = A@B + bias (store fp32), accumulate column sum/sumsq
// MODE 1: A = relu(z*scale+shift) (fused BN apply); out = relu(A@B + bias)
// Bt is f16, transposed: Bt[n][k].
template <int MODE>
__global__ __launch_bounds__(256) void gemm_kernel(const float* __restrict__ A,
                                                   const _Float16* __restrict__ Bt,
                                                   const float* __restrict__ bias,
                                                   float* __restrict__ out,
                                                   const float* __restrict__ scale,
                                                   const float* __restrict__ shift,
                                                   float* __restrict__ colsum,
                                                   float* __restrict__ colsq,
                                                   int nRows) {
    __shared__ __align__(16) _Float16 As[TM * LDA];
    __shared__ float sScale[HDIM], sShift[HDIM];

    const int tid   = threadIdx.x;
    const int mBase = blockIdx.x * TM;

    if (MODE == 1 && tid < HDIM) {
        sScale[tid] = scale[tid];
        sShift[tid] = shift[tid];
    }
    __syncthreads();

    // ---- stage A tile (128x128) into LDS as f16; thread owns a 64-col half-row
    {
        int r  = tid >> 1;               // row within tile
        int cb = (tid & 1) * 64;         // column base
        int row = mBase + r;
        bool inr = row < nRows;
        const float4* gp = (const float4*)&A[row * HDIM + cb];
        #pragma unroll
        for (int i = 0; i < 16; i++) {
            float4 v = inr ? gp[i] : make_float4(0.f, 0.f, 0.f, 0.f);
            if (MODE == 1) {
                int c = cb + i * 4;
                v.x = fmaxf(v.x * sScale[c + 0] + sShift[c + 0], 0.f);
                v.y = fmaxf(v.y * sScale[c + 1] + sShift[c + 1], 0.f);
                v.z = fmaxf(v.z * sScale[c + 2] + sShift[c + 2], 0.f);
                v.w = fmaxf(v.w * sScale[c + 3] + sShift[c + 3], 0.f);
            }
            f16x4 h4 = { (_Float16)v.x, (_Float16)v.y, (_Float16)v.z, (_Float16)v.w };
            *(f16x4*)&As[r * LDA + cb + i * 4] = h4;
        }
    }
    __syncthreads();

    // ---- WMMA: wave w computes rows [16w,16w+16) x all 128 cols
    const int wave = tid >> 5;
    const int lane = tid & 31;
    const int l16  = lane & 15;
    const int kg   = lane >> 4;                 // K-half select per ISA layout
    const int rowA = wave * 16 + l16;

    v8f acc[8] = {};

    #pragma unroll
    for (int kk = 0; kk < HDIM; kk += 32) {
        const _Float16* ap = &As[rowA * LDA + kk + kg * 8];
        f16x8 a0 = *(const f16x8*)ap;           // K = kk + kg*8 + [0,8)
        f16x8 a1 = *(const f16x8*)(ap + 16);    // K = kk + 16 + kg*8 + [0,8)
        v16h afrag = cat8(a0, a1);
        #pragma unroll
        for (int s = 0; s < 8; s++) {
            int n = s * 16 + l16;
            const _Float16* bp = &Bt[n * HDIM + kk + kg * 8];
            f16x8 b0 = *(const f16x8*)bp;
            f16x8 b1 = *(const f16x8*)(bp + 16);
            v16h bfrag = cat8(b0, b1);
            acc[s] = __builtin_amdgcn_wmma_f32_16x16x32_f16(
                false, afrag, false, bfrag, (short)0, acc[s], false, false);
        }
    }

    // ---- epilogue; C layout: VGPR r holds M = r + 8*kg, N = l16 (+ subtile base)
    #pragma unroll
    for (int s = 0; s < 8; s++) {
        int n = s * 16 + l16;
        float b = bias[n];
        float lsum = 0.f, lsq = 0.f;
        #pragma unroll
        for (int r = 0; r < 8; r++) {
            int row = mBase + wave * 16 + r + kg * 8;
            float v = acc[s][r] + b;
            if (MODE == 1) v = fmaxf(v, 0.f);
            if (row < nRows) {
                out[row * HDIM + n] = v;
                lsum += v;
                lsq  += v * v;
            }
        }
        if (MODE == 0) {
            atomicAddF32(&colsum[n], lsum);
            atomicAddF32(&colsq[n], lsq);
        }
    }
}

// segment_csr sum readout: block = graph, thread = feature (ptr is sorted)
__global__ __launch_bounds__(128) void readout(const float* __restrict__ h,
                                               const int* __restrict__ ptr,
                                               float* __restrict__ out, int G) {
    int g = blockIdx.x, f = threadIdx.x;
    int s = ptr[g], e = ptr[g + 1];
    float acc = 0.f;
    for (int n = s; n < e; n++) acc += h[n * HDIM + f];
    out[g * HDIM + f] = acc;
}

// ---------------------------------------------------------------------------
extern "C" void kernel_launch(void* const* d_in, const int* in_sizes, int n_in,
                              void* d_out, int out_size, void* d_ws, size_t ws_size,
                              hipStream_t stream) {
    const float* x    = (const float*)d_in[0];
    const int*   esrc = (const int*)d_in[1];
    const int*   edst = (const int*)d_in[2];
    const int*   ptr  = (const int*)d_in[3];
    const float* Wa[3]   = { (const float*)d_in[4],  (const float*)d_in[10], (const float*)d_in[16] };
    const float* ba[3]   = { (const float*)d_in[5],  (const float*)d_in[11], (const float*)d_in[17] };
    const float* gam[3]  = { (const float*)d_in[6],  (const float*)d_in[12], (const float*)d_in[18] };
    const float* bet[3]  = { (const float*)d_in[7],  (const float*)d_in[13], (const float*)d_in[19] };
    const float* Wb[3]   = { (const float*)d_in[8],  (const float*)d_in[14], (const float*)d_in[20] };
    const float* bb[3]   = { (const float*)d_in[9],  (const float*)d_in[15], (const float*)d_in[21] };

    const int N = in_sizes[0] / HDIM;   // 50000
    const int E = in_sizes[1];          // 800000
    const int G = in_sizes[3] - 1;      // 512

    char* ws = (char*)d_ws;
    size_t nb = (size_t)N * HDIM * sizeof(float);
    float*     hbuf = (float*)(ws);
    float*     agg  = (float*)(ws + nb);
    float*     zbuf = (float*)(ws + 2 * nb);
    _Float16*  wT   = (_Float16*)(ws + 3 * nb);
    float*     st   = (float*)(ws + 3 * nb + 6 * HDIM * HDIM * sizeof(_Float16));
    float* colsum = st;        float* colsq = st + 128;
    float* scale  = st + 256;  float* shift = st + 384;

    // weights -> f16 transposed, once per launch (deterministic)
    for (int l = 0; l < 3; l++) {
        prep_w<<<64, 256, 0, stream>>>(Wa[l], wT + (l * 2 + 0) * HDIM * HDIM);
        prep_w<<<64, 256, 0, stream>>>(Wb[l], wT + (l * 2 + 1) * HDIM * HDIM);
    }

    const int gemmBlocks = (N + TM - 1) / TM;
    const long long tE = (long long)E * 32;
    const unsigned edgeBlocks = (unsigned)((tE + 255) / 256);
    const int n4 = N * (HDIM / 4);

    const float* hin = x;
    for (int l = 0; l < 3; l++) {
        copy_kernel<<<(n4 + 255) / 256, 256, 0, stream>>>((float4*)agg, (const float4*)hin, n4);
        edge_scatter<<<edgeBlocks, 256, 0, stream>>>(agg, hin, esrc, edst, E);
        zero_stats<<<1, 256, 0, stream>>>(colsum);
        gemm_kernel<0><<<gemmBlocks, 256, 0, stream>>>(
            agg, wT + (l * 2 + 0) * HDIM * HDIM, ba[l], zbuf,
            nullptr, nullptr, colsum, colsq, N);
        bn_finalize<<<1, 128, 0, stream>>>(colsum, colsq, gam[l], bet[l], scale, shift, 1.0f / (float)N);
        gemm_kernel<1><<<gemmBlocks, 256, 0, stream>>>(
            zbuf, wT + (l * 2 + 1) * HDIM * HDIM, bb[l], hbuf,
            scale, shift, nullptr, nullptr, N);
        hin = hbuf;
    }
    readout<<<G, 128, 0, stream>>>(hbuf, ptr, (float*)d_out, G);
}
